// ViT_33303176413561
// MI455X (gfx1250) — matrix operand order repaired and checked by reading, not compiled, not run on hardware
//
#include <hip/hip_runtime.h>

typedef unsigned short u16;
typedef __bf16 bf16x16 __attribute__((ext_vector_type(16)));
typedef float  f32x8   __attribute__((ext_vector_type(8)));
typedef int    v4i     __attribute__((ext_vector_type(4)));

#define EDIM   768
#define FFDIM  3072
#define DEPTHN 12
#define BATCH  32
#define NSEQ   197
#define MPAD   6400     // padded B*NSEQ (=6304) to multiple of 128
#define MPATCH 6272     // B*196, already multiple of 128
#define NKP    224      // keys padded to 7*32 for WMMA contraction
#define HEADS  12
#define HDIM   64
#define NCLS   1000
#define NCLSP  1024

#define GLOBAL_AS __attribute__((address_space(1)))
#define LDS_AS    __attribute__((address_space(3)))

#if defined(__has_builtin)
#if __has_builtin(__builtin_amdgcn_global_load_async_to_lds_b128) && \
    __has_builtin(__builtin_amdgcn_s_wait_asynccnt)
#define USE_ASYNC_LDS 1
#endif
#endif

// ---------- helpers ----------

__device__ __forceinline__ u16 f2bf(float f) {
  unsigned u = __float_as_uint(f);
  unsigned r = u + 0x7FFFu + ((u >> 16) & 1u);   // round-to-nearest-even
  return (u16)(r >> 16);
}

union BF16Frag { bf16x16 v; uint4 q[2]; };

__device__ __forceinline__ bf16x16 ld_frag(const u16* p0, const u16* p1) {
  BF16Frag u;
  u.q[0] = *reinterpret_cast<const uint4*>(p0);
  u.q[1] = *reinterpret_cast<const uint4*>(p1);
  return u.v;
}

__device__ __forceinline__ f32x8 zero8() {
  f32x8 z;
#pragma unroll
  for (int i = 0; i < 8; ++i) z[i] = 0.f;
  return z;
}

__device__ __forceinline__ f32x8 wmma_bf16(bf16x16 a, bf16x16 b, f32x8 c) {
  // D = A(16x32 bf16) x B(32x16 bf16) + C(16x16 f32)
  return __builtin_amdgcn_wmma_f32_16x16x32_bf16(false, a, false, b,
                                                 (short)0, c, false, false);
}

#ifdef USE_ASYNC_LDS
__device__ __forceinline__ void async_cp16(const u16* g, u16* l) {
  __builtin_amdgcn_global_load_async_to_lds_b128(
      (GLOBAL_AS v4i*)g, (LDS_AS v4i*)l, 0, 0);
}
#endif

// ---------- GEMM: C[M,N] = A[M,K](bf16) * BT[N,K](bf16)^T  ----------
// A, BT row-major k-contiguous. 3-stage async LDS ring (ASYNCcnt-throttled,
// async loads retire in order so s_wait_asynccnt(4) == "oldest tile landed"),
// falling back to a register-staged double buffer + global_prefetch when the
// async builtins are unavailable.
#define BM 128
#define BN 128
#define BK 32
#define NBUF 3

__global__ __launch_bounds__(256) void gemm_bf16_kernel(
    const u16* __restrict__ A, const u16* __restrict__ BT,
    float* __restrict__ Cf, u16* __restrict__ Cb,
    const float* __restrict__ bias, const float* __restrict__ res,
    int M, int N, int K, int relu)
{
  __shared__ u16 lA[NBUF][BM * BK];
  __shared__ u16 lB[NBUF][BN * BK];

  const int tid  = threadIdx.x;
  const int lane = tid & 31;
  const int wave = tid >> 5;
  const int mBase = blockIdx.y * BM;
  const int nBase = blockIdx.x * BN;
  const int wM = (wave >> 2) * 64;   // 0 or 64
  const int wN = (wave & 3) * 32;    // 0,32,64,96

  f32x8 acc[4][2];
#pragma unroll
  for (int mi = 0; mi < 4; ++mi)
#pragma unroll
    for (int ni = 0; ni < 2; ++ni) acc[mi][ni] = zero8();

  const int lr = tid >> 1;           // 0..127: tile row
  const int lc = (tid & 1) * 16;     // 0 or 16: k-offset
  const u16* gA = A  + (size_t)(mBase + lr) * K + lc;
  const u16* gB = BT + (size_t)(nBase + lr) * K + lc;
  u16* sA = &lA[0][lr * BK + lc];    // stage 0 slot (stage s = +s*BM*BK)
  u16* sB = &lB[0][lr * BK + lc];

  const int rA     = lane & 15;
  const int kOffA  = (lane & 16) ? 16 : 0;   // A frag k-chunks {k0, k0+8}
  const int kOffB  = (lane & 16) ? 16 : 0;   // B frag k-chunks {k0, k0+8}
  const int nk = K / BK;
  int buf = 0;

#ifdef USE_ASYNC_LDS
  // ---- 3-deep async pipeline: tiles kt..kt+2 in flight, 1 barrier/step ----
  {
    const int pre = nk < 2 ? nk : 2;
    for (int s = 0; s < pre; ++s) {
      const u16* ga = gA + (size_t)s * BK;
      const u16* gb = gB + (size_t)s * BK;
      u16* da = sA + s * (BM * BK);
      u16* db = sB + s * (BN * BK);
      async_cp16(ga, da); async_cp16(ga + 8, da + 8);
      async_cp16(gb, db); async_cp16(gb + 8, db + 8);
    }
  }
  int nbuf = 2 < nk ? 2 : (nk - 1);  // ring slot for next issued tile
  for (int kt = 0; kt < nk; ++kt) {
    // oldest tile (kt) landed; tile kt+1 may still be in flight (4 ops/wave)
    if (kt + 1 < nk) __builtin_amdgcn_s_wait_asynccnt(4);
    else             __builtin_amdgcn_s_wait_asynccnt(0);
    __syncthreads();   // cross-wave visibility + everyone done with ring slot
    if (kt + 2 < nk) {
      const u16* ga = gA + (size_t)(kt + 2) * BK;
      const u16* gb = gB + (size_t)(kt + 2) * BK;
      u16* da = sA + nbuf * (BM * BK);
      u16* db = sB + nbuf * (BN * BK);
      async_cp16(ga, da); async_cp16(ga + 8, da + 8);
      async_cp16(gb, db); async_cp16(gb + 8, db + 8);
      nbuf = (nbuf == NBUF - 1) ? 0 : nbuf + 1;
    }
#else
  // ---- register-staged double buffer with prefetch ----
  uint4 ra0, ra1, rb0, rb1;
  {
    const uint4* ga = reinterpret_cast<const uint4*>(gA);
    ra0 = ga[0]; ra1 = ga[1];
    const uint4* gb = reinterpret_cast<const uint4*>(gB);
    rb0 = gb[0]; rb1 = gb[1];
  }
  {
    uint4* pa = reinterpret_cast<uint4*>(sA);
    pa[0] = ra0; pa[1] = ra1;
    uint4* pb = reinterpret_cast<uint4*>(sB);
    pb[0] = rb0; pb[1] = rb1;
  }
  __syncthreads();
  for (int kt = 0; kt < nk; ++kt) {
    if (kt + 1 < nk) {
      const uint4* ga = reinterpret_cast<const uint4*>(gA + (size_t)(kt + 1) * BK);
      ra0 = ga[0]; ra1 = ga[1];
      const uint4* gb = reinterpret_cast<const uint4*>(gB + (size_t)(kt + 1) * BK);
      rb0 = gb[0]; rb1 = gb[1];
      if (kt + 2 < nk) {  // confirmed: lowers to global_prefetch_b8
        __builtin_prefetch(gA + (size_t)(kt + 2) * BK, 0, 1);
        __builtin_prefetch(gB + (size_t)(kt + 2) * BK, 0, 1);
      }
    }
#endif

    // ---- compute from lA[buf]/lB[buf] ----
    {
      const u16* baseA = &lA[buf][0];
      const u16* baseB = &lB[buf][0];
      bf16x16 af[4], bfr[2];
#pragma unroll
      for (int mi = 0; mi < 4; ++mi) {
        const u16* p = baseA + (wM + mi * 16 + rA) * BK;
        af[mi] = ld_frag(p + kOffA, p + kOffA + 8);
      }
#pragma unroll
      for (int ni = 0; ni < 2; ++ni) {
        const u16* p = baseB + (wN + ni * 16 + rA) * BK;
        bfr[ni] = ld_frag(p + kOffB, p + kOffB + 8);
      }
#pragma unroll
      for (int mi = 0; mi < 4; ++mi)
#pragma unroll
        for (int ni = 0; ni < 2; ++ni)
          acc[mi][ni] = wmma_bf16(af[mi], bfr[ni], acc[mi][ni]);
    }

#ifdef USE_ASYNC_LDS
    buf = (buf == NBUF - 1) ? 0 : buf + 1;
  }
#else
    if (kt + 1 < nk) {
      __syncthreads();   // everyone done reading buf^1 from previous step
      const int nb = buf ^ 1;
      uint4* pa = reinterpret_cast<uint4*>(sA + nb * (BM * BK));
      pa[0] = ra0; pa[1] = ra1;
      uint4* pb = reinterpret_cast<uint4*>(sB + nb * (BN * BK));
      pb[0] = rb0; pb[1] = rb1;
      __syncthreads();
      buf = nb;
    }
  }
#endif

  // epilogue: acc layout — vgpr r, lane l: row = r + (l<16?0:8), col = l&15
  const int rowOff = (lane & 16) ? 8 : 0;
  const int col    = lane & 15;
#pragma unroll
  for (int mi = 0; mi < 4; ++mi) {
#pragma unroll
    for (int ni = 0; ni < 2; ++ni) {
      const int gc = nBase + wN + ni * 16 + col;
      float bv = bias ? bias[gc] : 0.f;
#pragma unroll
      for (int r = 0; r < 8; ++r) {
        const int gr = mBase + wM + mi * 16 + rowOff + r;
        float v = acc[mi][ni][r] + bv;
        if (res)  v += res[(size_t)gr * N + gc];
        if (relu) v = v > 0.f ? v : 0.f;
        if (Cf) Cf[(size_t)gr * N + gc] = v;
        if (Cb) Cb[(size_t)gr * N + gc] = f2bf(v);
      }
    }
  }
}

// ---------- attention: one workgroup (4 waves) per (batch, head) ----------
__global__ __launch_bounds__(128) void attn_kernel(
    const u16* __restrict__ Q, const u16* __restrict__ Kb,
    const u16* __restrict__ Vb, u16* __restrict__ O, float scale)
{
  extern __shared__ __align__(16) unsigned char smem_raw[];
  u16*   Kl = (u16*)smem_raw;            // [NKP][64]
  u16*   Vt = Kl + NKP * HDIM;           // [64][NKP] (V transposed)
  float* S  = (float*)(Vt + HDIM * NKP); // [4 waves][16][NKP]
  u16*   P  = (u16*)(S + 4 * 16 * NKP);  // [4 waves][16][NKP] bf16 probs

  const int tid  = threadIdx.x;
  const int lane = tid & 31;
  const int wave = tid >> 5;
  const int bh = blockIdx.x;
  const int b  = bh / HEADS;
  const int hh = bh % HEADS;
  const size_t baseOff = (size_t)b * NSEQ * EDIM + hh * HDIM;

  // stage K rows (zero-pad beyond NSEQ)
  for (int i = tid; i < NKP * (HDIM / 8); i += 128) {
    int row = i >> 3, c0 = (i & 7) << 3;
    uint4 val = {0u, 0u, 0u, 0u};
    if (row < NSEQ)
      val = *reinterpret_cast<const uint4*>(Kb + baseOff + (size_t)row * EDIM + c0);
    *reinterpret_cast<uint4*>(Kl + row * HDIM + c0) = val;
  }
  // stage V transposed: Vt[d][j]
  for (int i = tid; i < NKP * (HDIM / 8); i += 128) {
    int j = i >> 3, d0 = (i & 7) << 3;
    uint4 val = {0u, 0u, 0u, 0u};
    if (j < NSEQ)
      val = *reinterpret_cast<const uint4*>(Vb + baseOff + (size_t)j * EDIM + d0);
    u16 tmp[8];
    *reinterpret_cast<uint4*>(tmp) = val;
#pragma unroll
    for (int u = 0; u < 8; ++u) Vt[(d0 + u) * NKP + j] = tmp[u];
  }
  __syncthreads();

  float* Sw = S + wave * 16 * NKP;
  u16*   Pw = P + wave * 16 * NKP;
  const int nl    = lane & 15;
  const int kOffA = (lane & 16) ? 16 : 0;
  const int kOffB = (lane & 16) ? 16 : 0;
  const int rowOff = (lane & 16) ? 8 : 0;

  for (int qt = wave; qt < 13; qt += 4) {
    // --- load Q fragments straight from global (two k-steps over d=64) ---
    const int qrow = qt * 16 + nl;
    const bool qv = qrow < NSEQ;
    bf16x16 aq[2];
#pragma unroll
    for (int kk = 0; kk < 2; ++kk) {
      BF16Frag u;
      u.q[0] = {0u, 0u, 0u, 0u};
      u.q[1] = {0u, 0u, 0u, 0u};
      if (qv) {
        const u16* p = Q + baseOff + (size_t)qrow * EDIM + kk * 32 + kOffA;
        u.q[0] = *reinterpret_cast<const uint4*>(p);
        u.q[1] = *reinterpret_cast<const uint4*>(p + 8);
      }
      aq[kk] = u.v;
    }
    // --- S = scale * Q K^T over 14 key tiles ---
    for (int kt = 0; kt < 14; ++kt) {
      f32x8 acc = zero8();
      const u16* kb0 = Kl + (kt * 16 + nl) * HDIM;
      bf16x16 b0 = ld_frag(kb0 + kOffB, kb0 + kOffB + 8);
      bf16x16 b1 = ld_frag(kb0 + 32 + kOffB, kb0 + 32 + kOffB + 8);
      acc = wmma_bf16(aq[0], b0, acc);
      acc = wmma_bf16(aq[1], b1, acc);
#pragma unroll
      for (int r = 0; r < 8; ++r)
        Sw[(rowOff + r) * NKP + kt * 16 + nl] = acc[r] * scale;
    }
    // --- softmax per row (wave-cooperative, 224 = 7*32) ---
    for (int r = 0; r < 16; ++r) {
      float vals[7];
      float mx = -1e30f;
#pragma unroll
      for (int i = 0; i < 7; ++i) {
        int c = lane + i * 32;
        float v = (c < NSEQ) ? Sw[r * NKP + c] : -1e30f;
        vals[i] = v;
        mx = fmaxf(mx, v);
      }
#pragma unroll
      for (int o = 16; o > 0; o >>= 1) mx = fmaxf(mx, __shfl_xor(mx, o, 32));
      float sum = 0.f;
#pragma unroll
      for (int i = 0; i < 7; ++i) {
        int c = lane + i * 32;
        float p = (c < NSEQ) ? __expf(vals[i] - mx) : 0.f;
        vals[i] = p;
        sum += p;
      }
#pragma unroll
      for (int o = 16; o > 0; o >>= 1) sum += __shfl_xor(sum, o, 32);
      float inv = 1.f / sum;
#pragma unroll
      for (int i = 0; i < 7; ++i)
        Pw[r * NKP + lane + i * 32] = f2bf(vals[i] * inv);
    }
    // --- O = P V  (contraction over 224 keys = 7 k-steps, N = 64) ---
    f32x8 oacc[4];
#pragma unroll
    for (int ni = 0; ni < 4; ++ni) oacc[ni] = zero8();
    for (int ks = 0; ks < 7; ++ks) {
      const u16* pr = Pw + nl * NKP + ks * 32 + kOffA;
      bf16x16 ap = ld_frag(pr, pr + 8);
#pragma unroll
      for (int ni = 0; ni < 4; ++ni) {
        const u16* vb0 = Vt + (ni * 16 + nl) * NKP + ks * 32 + kOffB;
        bf16x16 bv = ld_frag(vb0, vb0 + 8);
        oacc[ni] = wmma_bf16(ap, bv, oacc[ni]);
      }
    }
#pragma unroll
    for (int ni = 0; ni < 4; ++ni) {
#pragma unroll
      for (int r = 0; r < 8; ++r) {
        int qr = qt * 16 + rowOff + r;
        if (qr < NSEQ)
          O[baseOff + (size_t)qr * EDIM + ni * 16 + nl] = f2bf(oacc[ni][r]);
      }
    }
  }
}

// ---------- LayerNorm: one wave per 768-wide row ----------
__global__ __launch_bounds__(256) void ln_kernel(
    const float* __restrict__ X, const float* __restrict__ g,
    const float* __restrict__ bta, float* __restrict__ Yf,
    u16* __restrict__ Yb, int rows)
{
  const int lane = threadIdx.x & 31, wave = threadIdx.x >> 5;
  const int row = blockIdx.x * 8 + wave;
  if (row >= rows) return;
  const float* xr = X + (size_t)row * EDIM;
  float v[24], s = 0.f, sq = 0.f;
#pragma unroll
  for (int i = 0; i < 24; ++i) {
    float x = xr[lane + i * 32];
    v[i] = x; s += x; sq += x * x;
  }
#pragma unroll
  for (int o = 16; o > 0; o >>= 1) {
    s  += __shfl_xor(s, o, 32);
    sq += __shfl_xor(sq, o, 32);
  }
  float m = s * (1.f / EDIM);
  float var = sq * (1.f / EDIM) - m * m;
  float rstd = rsqrtf(var + 1e-5f);
#pragma unroll
  for (int i = 0; i < 24; ++i) {
    int c = lane + i * 32;
    float y = (v[i] - m) * rstd * g[c] + bta[c];
    if (Yf) Yf[(size_t)row * EDIM + c] = y;
    if (Yb) Yb[(size_t)row * EDIM + c] = f2bf(y);
  }
}

// ---------- CLS extraction + final LN (rows padded to 128) ----------
__global__ __launch_bounds__(256) void cls_ln_kernel(
    const float* __restrict__ hf, const float* __restrict__ g,
    const float* __restrict__ bta, u16* __restrict__ out)
{
  const int lane = threadIdx.x & 31, wave = threadIdx.x >> 5;
  const int row = blockIdx.x * 8 + wave;
  if (row >= 128) return;
  if (row >= BATCH) {
#pragma unroll
    for (int i = 0; i < 24; ++i) out[(size_t)row * EDIM + lane + i * 32] = 0;
    return;
  }
  const float* xr = hf + (size_t)row * NSEQ * EDIM;  // token 0 of batch `row`
  float v[24], s = 0.f, sq = 0.f;
#pragma unroll
  for (int i = 0; i < 24; ++i) {
    float x = xr[lane + i * 32];
    v[i] = x; s += x; sq += x * x;
  }
#pragma unroll
  for (int o = 16; o > 0; o >>= 1) {
    s  += __shfl_xor(s, o, 32);
    sq += __shfl_xor(sq, o, 32);
  }
  float m = s * (1.f / EDIM);
  float var = sq * (1.f / EDIM) - m * m;
  float rstd = rsqrtf(var + 1e-5f);
#pragma unroll
  for (int i = 0; i < 24; ++i) {
    int c = lane + i * 32;
    out[(size_t)row * EDIM + c] = f2bf((v[i] - m) * rstd * g[c] + bta[c]);
  }
}

// ---------- weight transforms ----------
// dst[n][k] (bf16, k-contiguous) = src[k][n] (f32), zero-pad n >= Nsrc
__global__ void transpose_cast_kernel(const float* __restrict__ src,
                                      u16* __restrict__ dst,
                                      int K, int Nsrc, int Nr)
{
  int idx = blockIdx.x * 256 + threadIdx.x;
  if (idx >= Nr * K) return;
  int n = idx / K, k = idx - n * K;
  float v = (n < Nsrc) ? src[(size_t)k * Nsrc + n] : 0.f;
  dst[idx] = f2bf(v);
}

__global__ void cast_kernel(const float* __restrict__ src, u16* __restrict__ dst,
                            int count)
{
  int idx = blockIdx.x * 256 + threadIdx.x;
  if (idx < count) dst[idx] = f2bf(src[idx]);
}

// ---------- patch embedding pieces ----------
__global__ void im2col_kernel(const float* __restrict__ x, u16* __restrict__ patches)
{
  int idx = blockIdx.x * 256 + threadIdx.x;   // over 6272*768
  if (idx >= MPATCH * EDIM) return;
  int row = idx / EDIM, k = idx - row * EDIM;
  int b = row / 196, p = row - b * 196;
  int c = k >> 8, rem = k & 255;
  int i = rem >> 4, j = rem & 15;
  int py = p / 14, px = p - py * 14;
  float v = x[(((size_t)b * 3 + c) * 224 + (py * 16 + i)) * 224 + (px * 16 + j)];
  patches[idx] = f2bf(v);
}

__global__ void assemble_kernel(const float* __restrict__ pe,
                                const float* __restrict__ cls,
                                const float* __restrict__ pos,
                                float* __restrict__ hf, u16* __restrict__ hb)
{
  int idx = blockIdx.x * 256 + threadIdx.x;   // over B*197*768
  if (idx >= BATCH * NSEQ * EDIM) return;
  int c = idx % EDIM;
  int row = idx / EDIM;
  int b = row / NSEQ, n = row - b * NSEQ;
  float v;
  if (n == 0) v = cls[c] + pos[c];
  else        v = pe[((size_t)b * 196 + (n - 1)) * EDIM + c] + pos[(size_t)n * EDIM + c];
  hf[idx] = v;
  hb[idx] = f2bf(v);
}

__global__ void padbias_kernel(const float* __restrict__ src, float* __restrict__ dst)
{
  int i = blockIdx.x * 256 + threadIdx.x;
  if (i < NCLSP) dst[i] = (i < NCLS) ? src[i] : 0.f;
}

__global__ void copyout_kernel(const float* __restrict__ src, float* __restrict__ dst)
{
  int idx = blockIdx.x * 256 + threadIdx.x;   // over 32*1000
  if (idx >= BATCH * NCLS) return;
  int b = idx / NCLS, j = idx - b * NCLS;
  dst[idx] = src[(size_t)b * NCLSP + j];
}

// ---------- host driver ----------

static inline void launch_gemm(const u16* A, const u16* BT, float* Cf, u16* Cb,
                               const float* bias, const float* res,
                               int M, int N, int K, int relu, hipStream_t s)
{
  dim3 grid(N / BN, M / BM);
  gemm_bf16_kernel<<<grid, 256, 0, s>>>(A, BT, Cf, Cb, bias, res, M, N, K, relu);
}

extern "C" void kernel_launch(void* const* d_in, const int* in_sizes, int n_in,
                              void* d_out, int out_size, void* d_ws, size_t ws_size,
                              hipStream_t stream)
{
  (void)in_sizes; (void)n_in; (void)out_size; (void)ws_size;
  const float* x        = (const float*)d_in[0];
  const float* conv_w   = (const float*)d_in[1];
  const float* conv_b   = (const float*)d_in[2];
  const float* cls_tok  = (const float*)d_in[3];
  const float* pos_emb  = (const float*)d_in[4];
  const float* Wq       = (const float*)d_in[5];
  const float* Wk       = (const float*)d_in[6];
  const float* Wv       = (const float*)d_in[7];
  const float* Wo       = (const float*)d_in[8];
  const float* bo       = (const float*)d_in[9];
  const float* ln1_g    = (const float*)d_in[10];
  const float* ln1_b    = (const float*)d_in[11];
  const float* W1       = (const float*)d_in[12];
  const float* b1       = (const float*)d_in[13];
  const float* W2       = (const float*)d_in[14];
  const float* b2       = (const float*)d_in[15];
  const float* ln2_g    = (const float*)d_in[16];
  const float* ln2_b    = (const float*)d_in[17];
  const float* head_g   = (const float*)d_in[18];
  const float* head_b   = (const float*)d_in[19];
  const float* head_W   = (const float*)d_in[20];
  const float* head_bias= (const float*)d_in[21];

  char* ws = (char*)d_ws;
  size_t off = 0;
  auto alloc = [&](size_t bytes) -> char* {
    char* p = ws + off;
    off += (bytes + 255) & ~(size_t)255;
    return p;
  };

  u16*   wT   = (u16*)  alloc((size_t)FFDIM * EDIM * 2);  // reused per weight
  u16*   hdT  = (u16*)  alloc((size_t)NCLSP * EDIM * 2);
  float* hbia = (float*)alloc((size_t)NCLSP * 4);
  float* hf   = (float*)alloc((size_t)MPAD * EDIM * 4);
  u16*   hb   = (u16*)  alloc((size_t)MPAD * EDIM * 2);
  u16*   qb   = (u16*)  alloc((size_t)MPAD * EDIM * 2);
  u16*   kb   = (u16*)  alloc((size_t)MPAD * EDIM * 2);
  u16*   vb   = (u16*)  alloc((size_t)MPAD * EDIM * 2);
  u16*   ab   = (u16*)  alloc((size_t)MPAD * EDIM * 2);
  float* tf   = (float*)alloc((size_t)MPAD * EDIM * 4);
  float* xf   = (float*)alloc((size_t)MPAD * EDIM * 4);
  u16*   xb   = (u16*)  alloc((size_t)MPAD * EDIM * 2);
  u16*   f1b  = (u16*)  alloc((size_t)MPAD * FFDIM * 2);
  u16*   clsb = (u16*)  alloc((size_t)128 * EDIM * 2);
  float* hout = (float*)alloc((size_t)128 * NCLSP * 4);
  u16*   patches = f1b;   // alias: patches used before f1b

  const int ROWS = BATCH * NSEQ;        // 6304
  const float scale = 1.0f / sqrtf((float)EDIM);
  const size_t attnSmem = (size_t)(2 * NKP * HDIM) * 2   // Kl + Vt (bf16)
                        + (size_t)(4 * 16 * NKP) * 4     // S (f32)
                        + (size_t)(4 * 16 * NKP) * 2;    // P (bf16)

  // ---- patch embedding ----
  cast_kernel<<<(EDIM * EDIM + 255) / 256, 256, 0, stream>>>(conv_w, wT, EDIM * EDIM);
  im2col_kernel<<<(MPATCH * EDIM + 255) / 256, 256, 0, stream>>>(x, patches);
  launch_gemm(patches, wT, tf, nullptr, conv_b, nullptr, MPATCH, EDIM, EDIM, 0, stream);
  assemble_kernel<<<(ROWS * EDIM + 255) / 256, 256, 0, stream>>>(tf, cls_tok, pos_emb, hf, hb);

  // ---- transformer blocks ----
  const int trBlkEE = (EDIM * EDIM + 255) / 256;
  const int trBlkEF = (EDIM * FFDIM + 255) / 256;
  for (int l = 0; l < DEPTHN; ++l) {
    const size_t oEE = (size_t)l * EDIM * EDIM;
    const size_t oEF = (size_t)l * EDIM * FFDIM;

    transpose_cast_kernel<<<trBlkEE, 256, 0, stream>>>(Wq + oEE, wT, EDIM, EDIM, EDIM);
    launch_gemm(hb, wT, nullptr, qb, nullptr, nullptr, MPAD, EDIM, EDIM, 0, stream);
    transpose_cast_kernel<<<trBlkEE, 256, 0, stream>>>(Wk + oEE, wT, EDIM, EDIM, EDIM);
    launch_gemm(hb, wT, nullptr, kb, nullptr, nullptr, MPAD, EDIM, EDIM, 0, stream);
    transpose_cast_kernel<<<trBlkEE, 256, 0, stream>>>(Wv + oEE, wT, EDIM, EDIM, EDIM);
    launch_gemm(hb, wT, nullptr, vb, nullptr, nullptr, MPAD, EDIM, EDIM, 0, stream);

    attn_kernel<<<BATCH * HEADS, 128, attnSmem, stream>>>(qb, kb, vb, ab, scale);

    transpose_cast_kernel<<<trBlkEE, 256, 0, stream>>>(Wo + oEE, wT, EDIM, EDIM, EDIM);
    launch_gemm(ab, wT, tf, nullptr, bo + (size_t)l * EDIM, hf, MPAD, EDIM, EDIM, 0, stream);
    ln_kernel<<<(ROWS + 7) / 8, 256, 0, stream>>>(tf, ln1_g + (size_t)l * EDIM,
                                                  ln1_b + (size_t)l * EDIM, xf, xb, ROWS);

    transpose_cast_kernel<<<trBlkEF, 256, 0, stream>>>(W1 + oEF, wT, EDIM, FFDIM, FFDIM);
    launch_gemm(xb, wT, nullptr, f1b, b1 + (size_t)l * FFDIM, nullptr, MPAD, FFDIM, EDIM, 1, stream);
    transpose_cast_kernel<<<trBlkEF, 256, 0, stream>>>(W2 + oEF, wT, FFDIM, EDIM, EDIM);
    launch_gemm(f1b, wT, tf, nullptr, b2 + (size_t)l * EDIM, xf, MPAD, EDIM, FFDIM, 0, stream);
    ln_kernel<<<(ROWS + 7) / 8, 256, 0, stream>>>(tf, ln2_g + (size_t)l * EDIM,
                                                  ln2_b + (size_t)l * EDIM, hf, hb, ROWS);
  }

  // ---- head ----
  transpose_cast_kernel<<<(NCLSP * EDIM + 255) / 256, 256, 0, stream>>>(
      head_W, hdT, EDIM, NCLS, NCLSP);
  padbias_kernel<<<(NCLSP + 255) / 256, 256, 0, stream>>>(head_bias, hbia);
  cls_ln_kernel<<<16, 256, 0, stream>>>(hf, head_g, head_b, clsb);
  launch_gemm(clsb, hdT, hout, nullptr, hbia, nullptr, 128, NCLSP, EDIM, 0, stream);
  copyout_kernel<<<(BATCH * NCLS + 255) / 256, 256, 0, stream>>>(hout, (float*)d_out);
}
